// FuzzyMealy_13022340842105
// MI455X (gfx1250) — matrix operand order repaired and checked by reading, not compiled, not run on hardware
//
#include <hip/hip_runtime.h>

// ---------------------------------------------------------------------------
// FuzzyMealy on MI455X (gfx1250, wave32, WMMA bf16)
//
// state_{t+1} = state_t @ T[:,inp_t,:]  is a LINEAR recurrence -> chunked scan:
//   P0: softmax rows (max/sum stats), then normalize+transpose -> bf16 tables
//       Ttr[i][t][s], Otr[i][o][s]  (B-fragment-friendly layout, L2 resident)
//   P2a: per chunk c (256 chunks x 32 steps): M_c = T_{i0} @ ... @ T_{i31}
//        via sequential bf16 WMMA GEMMs, chunks parallel across WGPs
//   P2b: one block sequentially scans the 256 chunk matrices -> boundary states
//        (with global_prefetch of the next chunk matrix)
//   P2c: 256 parallel blocks recompute the 32 states inside each chunk (VALU)
//   P3 : group timesteps by symbol; per symbol GEMM States_i @ O_i via WMMA
// ---------------------------------------------------------------------------

#define NSTATE   512
#define NSYM     256
#define SEQLEN   8192
#define L_CHUNK  32
#define NCHUNKS  (SEQLEN / L_CHUNK)   // 256

typedef __bf16 bf16_t;
typedef __attribute__((ext_vector_type(8)))  __bf16 v8bf;
typedef __attribute__((ext_vector_type(16))) __bf16 v16bf;
typedef __attribute__((ext_vector_type(8)))  float  v8f;

// ---------------------------------------------------------------------------
// P0a: per-row (length 512) max and 1/sum(exp) ; one wave32 per row
// rows are (s,i) pairs of logits[s][i][512]
// ---------------------------------------------------------------------------
__global__ __launch_bounds__(256) void row_stats_kernel(
    const float* __restrict__ logits, float2* __restrict__ stats, int nrows)
{
    int wave = (int)(blockIdx.x * (blockDim.x / 32) + threadIdx.x / 32);
    int lane = (int)(threadIdx.x & 31);
    if (wave >= nrows) return;
    const float4* row = (const float4*)(logits + (size_t)wave * NSTATE);
    float mx = -3.0e38f;
    float4 t0 = row[0 * 32 + lane], t1 = row[1 * 32 + lane];
    float4 t2 = row[2 * 32 + lane], t3 = row[3 * 32 + lane];
    mx = fmaxf(mx, fmaxf(fmaxf(t0.x, t0.y), fmaxf(t0.z, t0.w)));
    mx = fmaxf(mx, fmaxf(fmaxf(t1.x, t1.y), fmaxf(t1.z, t1.w)));
    mx = fmaxf(mx, fmaxf(fmaxf(t2.x, t2.y), fmaxf(t2.z, t2.w)));
    mx = fmaxf(mx, fmaxf(fmaxf(t3.x, t3.y), fmaxf(t3.z, t3.w)));
    #pragma unroll
    for (int o = 16; o > 0; o >>= 1) mx = fmaxf(mx, __shfl_xor(mx, o, 32));
    float s = 0.f;
    s += __expf(t0.x - mx) + __expf(t0.y - mx) + __expf(t0.z - mx) + __expf(t0.w - mx);
    s += __expf(t1.x - mx) + __expf(t1.y - mx) + __expf(t1.z - mx) + __expf(t1.w - mx);
    s += __expf(t2.x - mx) + __expf(t2.y - mx) + __expf(t2.z - mx) + __expf(t2.w - mx);
    s += __expf(t3.x - mx) + __expf(t3.y - mx) + __expf(t3.z - mx) + __expf(t3.w - mx);
    #pragma unroll
    for (int o = 16; o > 0; o >>= 1) s += __shfl_xor(s, o, 32);
    if (lane == 0) stats[wave] = make_float2(mx, 1.f / s);
}

// ---------------------------------------------------------------------------
// P0b: normalize + transpose + bf16:  out_tr[i][n][s] = softmax(logits)[s][i][n]
// grid (i=256, s-tile=16, n-tile=4), block 256
// ---------------------------------------------------------------------------
__global__ __launch_bounds__(256) void softmax_transpose_kernel(
    const float* __restrict__ logits, const float2* __restrict__ stats,
    bf16_t* __restrict__ out_tr)
{
    int i  = (int)blockIdx.x;
    int s0 = (int)blockIdx.y * 32;
    int n0 = (int)blockIdx.z * 128;
    __shared__ bf16_t tile[32][129];
    int tid = (int)threadIdx.x;
    int r   = tid / 8;           // 32 s-rows
    int cb  = (tid % 8) * 16;    // 16 n each
    size_t rowidx = (size_t)(s0 + r) * NSYM + i;
    float2 st = stats[rowidx];
    const float* src = logits + rowidx * NSTATE + n0 + cb;
    #pragma unroll
    for (int q = 0; q < 16; ++q)
        tile[r][cb + q] = (bf16_t)(__expf(src[q] - st.x) * st.y);
    __syncthreads();
    int nn = tid / 2;            // 128 n-rows
    int sb = (tid % 2) * 16;     // 16 s each
    bf16_t* dst = out_tr + ((size_t)i * NSTATE + n0 + nn) * NSTATE + s0 + sb;
    #pragma unroll
    for (int q = 0; q < 16; ++q) dst[q] = tile[sb + q][nn];
}

// ---------------------------------------------------------------------------
// WMMA fragment helpers (wave32, v_wmma_f32_16x16x32_bf16)
// A 16x32:  lane L -> row m=L%16 ; elems e = K[(L/16)*16 + e]  (contig 32B)
// B 32x16:  lane L -> col n=L%16 ; elems e = K[(L/16)*16 + e]  (contig 32B)
// C/D f32:  lane L, elem e -> m = e + 8*(L/16), n = L%16
// ---------------------------------------------------------------------------
__device__ __forceinline__ v16bf load_a_frag(const bf16_t* rowp, int k0, int lh) {
    v16bf a;
    *(v8bf*)(&a)       = *(const v8bf*)(rowp + k0 + lh * 16);
    *((v8bf*)(&a) + 1) = *(const v8bf*)(rowp + k0 + lh * 16 + 8);
    return a;
}
__device__ __forceinline__ v16bf load_b_frag(const bf16_t* mat_tr, int n, int k0, int lh) {
    return *(const v16bf*)(mat_tr + (size_t)n * NSTATE + k0 + lh * 16);
}

// ---------------------------------------------------------------------------
// P2a: per-chunk matrix composition. 1 block per chunk, 256 threads (8 waves).
// bufA/bufB are per-chunk 512x512 bf16 ping-pong product buffers.
// Final composed matrix ends in bufB (L_CHUNK-1 = 31 matmuls, odd).
// ---------------------------------------------------------------------------
__global__ __launch_bounds__(256) void chunk_products_kernel(
    const bf16_t* __restrict__ Ttr, const int* __restrict__ seq,
    bf16_t* __restrict__ bufA, bf16_t* __restrict__ bufB)
{
    const int c   = (int)blockIdx.x;
    const int tid = (int)threadIdx.x;
    bf16_t* A = bufA + (size_t)c * NSTATE * NSTATE;
    bf16_t* B = bufB + (size_t)c * NSTATE * NSTATE;

    // --- init: A = transpose(Ttr[i0]) = T_{i0} row-major ---
    {
        __shared__ bf16_t tile[32][33];
        const int i0 = seq[c * L_CHUNK];
        const bf16_t* src = Ttr + (size_t)i0 * NSTATE * NSTATE;
        int r = tid / 8, cq = (tid % 8) * 4;
        for (int tb = 0; tb < 256; ++tb) {
            int t0 = (tb / 16) * 32, s0 = (tb % 16) * 32;
            const bf16_t* p = src + (size_t)(t0 + r) * NSTATE + s0 + cq;
            tile[r][cq + 0] = p[0]; tile[r][cq + 1] = p[1];
            tile[r][cq + 2] = p[2]; tile[r][cq + 3] = p[3];
            __syncthreads();
            bf16_t* q = A + (size_t)(s0 + r) * NSTATE + t0 + cq;
            q[0] = tile[cq + 0][r]; q[1] = tile[cq + 1][r];
            q[2] = tile[cq + 2][r]; q[3] = tile[cq + 3][r];
            __syncthreads();
        }
    }
    __threadfence();
    __syncthreads();

    const int wave = tid / 32, lane = tid & 31;
    const int wm = wave / 2, wn = wave & 1;    // 4 x 2 wave grid; wave tile 32x64
    const int lm = lane & 15, lh = lane >> 4;

    for (int j = 1; j < L_CHUNK; ++j) {
        const bf16_t* Asrc = (j & 1) ? A : B;
        bf16_t*       Dst  = (j & 1) ? B : A;
        const bf16_t* Bsrc = Ttr + (size_t)seq[c * L_CHUNK + j] * NSTATE * NSTATE;

        for (int tb = 0; tb < 16; ++tb) {          // 4x4 grid of 128x128 WG tiles
            int m0 = (tb / 4) * 128 + wm * 32;
            int n0 = (tb % 4) * 128 + wn * 64;
            v8f acc[2][4];
            #pragma unroll
            for (int a_ = 0; a_ < 2; ++a_)
                #pragma unroll
                for (int b_ = 0; b_ < 4; ++b_)
                    #pragma unroll
                    for (int e = 0; e < 8; ++e) acc[a_][b_][e] = 0.f;

            for (int k0 = 0; k0 < NSTATE; k0 += 32) {
                v16bf af[2], bfv[4];
                #pragma unroll
                for (int mi = 0; mi < 2; ++mi)
                    af[mi] = load_a_frag(Asrc + (size_t)(m0 + mi * 16 + lm) * NSTATE, k0, lh);
                #pragma unroll
                for (int ni = 0; ni < 4; ++ni)
                    bfv[ni] = load_b_frag(Bsrc, n0 + ni * 16 + lm, k0, lh);
                #pragma unroll
                for (int mi = 0; mi < 2; ++mi)
                    #pragma unroll
                    for (int ni = 0; ni < 4; ++ni)
                        acc[mi][ni] = __builtin_amdgcn_wmma_f32_16x16x32_bf16(
                            false, af[mi], false, bfv[ni], (short)0, acc[mi][ni], false, false);
            }
            #pragma unroll
            for (int mi = 0; mi < 2; ++mi)
                #pragma unroll
                for (int ni = 0; ni < 4; ++ni)
                    #pragma unroll
                    for (int e = 0; e < 8; ++e) {
                        int m = m0 + mi * 16 + lh * 8 + e;
                        int n = n0 + ni * 16 + lm;
                        Dst[(size_t)m * NSTATE + n] = (bf16_t)acc[mi][ni][e];
                    }
        }
        __threadfence();
        __syncthreads();
    }
}

// ---------------------------------------------------------------------------
// P2b: sequential scan over 256 chunk matrices (single block, 512 threads)
// boundaries[c] = state at the START of chunk c.  Prefetch next chunk matrix.
// ---------------------------------------------------------------------------
__global__ __launch_bounds__(512) void chunk_scan_kernel(
    const bf16_t* __restrict__ Mfinal, const int* __restrict__ init_idx,
    float* __restrict__ boundaries)
{
    __shared__ float st[NSTATE];
    int tid = (int)threadIdx.x;
    st[tid] = (tid == *init_idx) ? 1.f : 0.f;
    __syncthreads();
    for (int c = 0; c < NCHUNKS; ++c) {
        boundaries[(size_t)c * NSTATE + tid] = st[tid];
        const bf16_t* M = Mfinal + (size_t)c * NSTATE * NSTATE;
        // prefetch next chunk's matrix (512 threads x 16 rows -> all cachelines)
        if (c + 1 < NCHUNKS) {
            const bf16_t* Mn = M + (size_t)NSTATE * NSTATE;
            #pragma unroll
            for (int s = 0; s < NSTATE; s += 32)
                __builtin_prefetch(Mn + (size_t)(s + (tid >> 4)) * NSTATE + (tid & 15) * 32, 0, 1);
        }
        float acc = 0.f;
        #pragma unroll 4
        for (int s = 0; s < NSTATE; ++s)
            acc += st[s] * (float)M[(size_t)s * NSTATE + tid];
        __syncthreads();
        st[tid] = acc;
        __syncthreads();
    }
}

// ---------------------------------------------------------------------------
// P2c: parallel within-chunk state recomputation; states_bf[t][s] (bf16)
// one block per chunk, 512 threads (thread = state component)
// ---------------------------------------------------------------------------
__global__ __launch_bounds__(512) void recompute_states_kernel(
    const bf16_t* __restrict__ Ttr, const int* __restrict__ seq,
    const float* __restrict__ boundaries, bf16_t* __restrict__ states_bf)
{
    int c = (int)blockIdx.x, tid = (int)threadIdx.x;
    __shared__ float st[NSTATE];
    st[tid] = boundaries[(size_t)c * NSTATE + tid];
    __syncthreads();
    for (int j = 0; j < L_CHUNK; ++j) {
        int g = c * L_CHUNK + j;
        states_bf[(size_t)g * NSTATE + tid] = (bf16_t)st[tid];
        const bf16_t* Trow =
            Ttr + (size_t)seq[g] * NSTATE * NSTATE + (size_t)tid * NSTATE;
        float acc = 0.f;
        for (int s = 0; s < NSTATE; s += 8) {
            v8bf tv = *(const v8bf*)(Trow + s);
            #pragma unroll
            for (int q = 0; q < 8; ++q) acc += (float)tv[q] * st[s + q];
        }
        __syncthreads();
        st[tid] = acc;
        __syncthreads();
    }
}

// ---------------------------------------------------------------------------
// P3 prep: group timesteps by symbol
// ---------------------------------------------------------------------------
__global__ void zero_counts_kernel(int* counts) { counts[threadIdx.x] = 0; }

__global__ void hist_kernel(const int* __restrict__ seq, int* counts) {
    int t = (int)(blockIdx.x * 256 + threadIdx.x);
    atomicAdd(&counts[seq[t]], 1);
}
__global__ void scan_kernel(const int* __restrict__ counts, int* offsets, int* cursor) {
    __shared__ int tmp[NSYM + 1];
    int tid = (int)threadIdx.x;
    tmp[tid + 1] = counts[tid];
    if (tid == 0) tmp[0] = 0;
    __syncthreads();
    if (tid == 0)
        for (int q = 1; q <= NSYM; ++q) tmp[q] += tmp[q - 1];
    __syncthreads();
    offsets[tid] = tmp[tid];
    cursor[tid]  = tmp[tid];
    if (tid == NSYM - 1) offsets[NSYM] = tmp[NSYM];
}
__global__ void scatter_kernel(const int* __restrict__ seq, int* cursor, int* perm) {
    int t = (int)(blockIdx.x * 256 + threadIdx.x);
    int p = atomicAdd(&cursor[seq[t]], 1);
    perm[p] = t;
}

// ---------------------------------------------------------------------------
// P3: grouped output GEMM: out[perm rows] = States_i @ O_i, WMMA bf16
// one block per symbol, 8 waves, each wave covers a 64-wide N stripe
// ---------------------------------------------------------------------------
__global__ __launch_bounds__(256) void out_gemm_kernel(
    const bf16_t* __restrict__ states_bf, const bf16_t* __restrict__ Otr,
    const int* __restrict__ offsets, const int* __restrict__ perm,
    float* __restrict__ out)
{
    int i   = (int)blockIdx.x;
    int beg = offsets[i], cnt = offsets[i + 1] - beg;
    if (cnt <= 0) return;
    int wave = (int)threadIdx.x / 32, lane = (int)threadIdx.x & 31;
    int lm = lane & 15, lh = lane >> 4;
    int n0 = wave * 64;
    const bf16_t* Bsrc = Otr + (size_t)i * NSTATE * NSTATE;

    for (int mt = 0; mt < cnt; mt += 16) {
        int ra = mt + lm;
        const bf16_t* arow =
            (ra < cnt) ? (states_bf + (size_t)perm[beg + ra] * NSTATE) : nullptr;
        v8f acc[4];
        #pragma unroll
        for (int ni = 0; ni < 4; ++ni)
            #pragma unroll
            for (int e = 0; e < 8; ++e) acc[ni][e] = 0.f;

        for (int k0 = 0; k0 < NSTATE; k0 += 32) {
            v16bf af;
            if (arow) af = load_a_frag(arow, k0, lh);
            else {
                #pragma unroll
                for (int q = 0; q < 16; ++q) af[q] = (bf16_t)0.0f;
            }
            v16bf bfv[4];
            #pragma unroll
            for (int ni = 0; ni < 4; ++ni)
                bfv[ni] = load_b_frag(Bsrc, n0 + ni * 16 + lm, k0, lh);
            #pragma unroll
            for (int ni = 0; ni < 4; ++ni)
                acc[ni] = __builtin_amdgcn_wmma_f32_16x16x32_bf16(
                    false, af, false, bfv[ni], (short)0, acc[ni], false, false);
        }
        #pragma unroll
        for (int ni = 0; ni < 4; ++ni)
            #pragma unroll
            for (int e = 0; e < 8; ++e) {
                int m = mt + lh * 8 + e;
                if (m < cnt) {
                    int t = perm[beg + m];
                    out[(size_t)t * NSTATE + n0 + ni * 16 + lm] = acc[ni][e];
                }
            }
    }
}

// ---------------------------------------------------------------------------
// Host launcher.  Workspace budget ~550 MB (device has 432 GiB).
// ---------------------------------------------------------------------------
extern "C" void kernel_launch(void* const* d_in, const int* in_sizes, int n_in,
                              void* d_out, int out_size, void* d_ws, size_t ws_size,
                              hipStream_t stream) {
    const float* T_logits = (const float*)d_in[0];
    const float* O_logits = (const float*)d_in[1];
    const int*   init_idx = (const int*)d_in[2];
    const int*   seq      = (const int*)d_in[3];
    float*       out      = (float*)d_out;

    const size_t TBL = (size_t)NSYM * NSTATE * NSTATE * sizeof(bf16_t); // 128 MiB
    char* ws = (char*)d_ws;
    size_t off = 0;
    auto take = [&](size_t bytes) -> char* {
        char* p = ws + off;
        off = (off + bytes + 255) & ~(size_t)255;
        return p;
    };
    bf16_t* Ttr    = (bf16_t*)take(TBL);
    bf16_t* Otr    = (bf16_t*)take(TBL);
    bf16_t* bufA   = (bf16_t*)take(TBL);
    bf16_t* bufB   = (bf16_t*)take(TBL);
    bf16_t* states = (bf16_t*)take((size_t)SEQLEN * NSTATE * sizeof(bf16_t));
    float*  bounds = (float*)take((size_t)NCHUNKS * NSTATE * sizeof(float));
    float2* statsT = (float2*)take((size_t)NSTATE * NSYM * sizeof(float2));
    float2* statsO = (float2*)take((size_t)NSTATE * NSYM * sizeof(float2));
    int*    counts = (int*)take(NSYM * sizeof(int));
    int*    offs   = (int*)take((NSYM + 1) * sizeof(int));
    int*    cursor = (int*)take(NSYM * sizeof(int));
    int*    perm   = (int*)take(SEQLEN * sizeof(int));
    (void)ws_size; (void)n_in; (void)in_sizes; (void)out_size;

    const int nrows = NSTATE * NSYM; // 131072 softmax rows per table

    // P0: softmax stats + normalized bf16 transposed tables
    row_stats_kernel<<<dim3(nrows / 8), dim3(256), 0, stream>>>(T_logits, statsT, nrows);
    row_stats_kernel<<<dim3(nrows / 8), dim3(256), 0, stream>>>(O_logits, statsO, nrows);
    softmax_transpose_kernel<<<dim3(NSYM, 16, 4), dim3(256), 0, stream>>>(T_logits, statsT, Ttr);
    softmax_transpose_kernel<<<dim3(NSYM, 16, 4), dim3(256), 0, stream>>>(O_logits, statsO, Otr);

    // P2a: chunk transition-matrix composition (WMMA-heavy, 256 parallel chunks)
    chunk_products_kernel<<<dim3(NCHUNKS), dim3(256), 0, stream>>>(Ttr, seq, bufA, bufB);

    // P2b: sequential scan over composed chunk matrices -> boundary states
    chunk_scan_kernel<<<dim3(1), dim3(512), 0, stream>>>(bufB, init_idx, bounds);

    // P2c: parallel within-chunk state recomputation -> bf16 state rows
    recompute_states_kernel<<<dim3(NCHUNKS), dim3(512), 0, stream>>>(Ttr, seq, bounds, states);

    // P3: group timesteps by symbol, then grouped output GEMM
    zero_counts_kernel<<<dim3(1), dim3(NSYM), 0, stream>>>(counts);
    hist_kernel<<<dim3(SEQLEN / 256), dim3(256), 0, stream>>>(seq, counts);
    scan_kernel<<<dim3(1), dim3(NSYM), 0, stream>>>(counts, offs, cursor);
    scatter_kernel<<<dim3(SEQLEN / 256), dim3(256), 0, stream>>>(seq, cursor, perm);
    out_gemm_kernel<<<dim3(NSYM), dim3(256), 0, stream>>>(states, Otr, offs, perm, out);
}
